// CrystalGCN_17575006175633
// MI455X (gfx1250) — compile-verified, matrix operand order checked by reading, not run on hardware
//
#include <hip/hip_runtime.h>
#include <hip/hip_bf16.h>

// ---------------- problem constants ----------------
#define N_NODES  32000
#define N_EDGES  320000
#define N_GRAPHS 1600
#define HID      128
#define RBF      32
#define ZDIM     288            // 2*HID + RBF
#define KSTEPS   9              // 288 / 32
#define E_TILE   64             // edges per block
#define LDS_STRIDE 304          // 288 padded to 32B multiple (bank-spread)
#define FRAG_ELEMS 36864        // 8 ntiles * 9 ksteps * 32 lanes * 16 = 288*128

typedef __attribute__((ext_vector_type(16))) __bf16 v16bf;
typedef __attribute__((ext_vector_type(8)))  __bf16 v8bf;
typedef __attribute__((ext_vector_type(8)))  float  v8f;

__device__ __forceinline__ unsigned short f2bf(float f) {
    unsigned u = __float_as_uint(f);
    unsigned r = (u + 0x7FFFu + ((u >> 16) & 1u)) >> 16;   // round-to-nearest-even
    return (unsigned short)r;
}

// ---------------- weight -> bf16 WMMA B-fragment layout ----------------
// frag index t = (((ntile*9 + kstep)*32 + lane)*16 + i)
// element i of lane L holds W[kstep*32 + koff][ntile*16 + (L&15)]
// koff = ((L>>4)&1)*8 + (i<8 ? i : i+8)   (mirrors ISA 16-bit 16x32 fragment layout)
__global__ __launch_bounds__(256) void wfrag_kernel(const float* __restrict__ W,
                                                    unsigned short* __restrict__ o) {
    int t = blockIdx.x * 256 + threadIdx.x;
    if (t >= FRAG_ELEMS) return;
    int nt = t / 4608;          // 9*32*16
    int r  = t - nt * 4608;
    int ks = r >> 9;            // /512
    int r2 = r & 511;
    int L  = r2 >> 4;
    int i  = r2 & 15;
    int koff = ((L >> 4) & 1) * 8 + (i < 8 ? i : i + 8);
    int k = ks * 32 + koff;
    int n = nt * 16 + (L & 15);
    o[t] = f2bf(W[k * HID + n]);
}

// ---------------- elementwise helpers ----------------
__global__ __launch_bounds__(256) void embed_kernel(const int* __restrict__ x,
                                                    const float* __restrict__ emb,
                                                    float* __restrict__ h) {
    int t = blockIdx.x * 256 + threadIdx.x;          // < N_NODES*HID
    int n = t >> 7, c = t & 127;
    h[t] = emb[x[n] * HID + c];
}

__global__ __launch_bounds__(256) void copy_kernel(const float* __restrict__ a,
                                                   float* __restrict__ b) {
    int t = blockIdx.x * 256 + threadIdx.x;
    b[t] = a[t];
}

__global__ __launch_bounds__(256) void relu_kernel(float* __restrict__ a) {
    int t = blockIdx.x * 256 + threadIdx.x;
    a[t] = fmaxf(a[t], 0.0f);
}

__global__ __launch_bounds__(256) void zero_kernel(float* __restrict__ a, int n) {
    int t = blockIdx.x * 256 + threadIdx.x;
    if (t < n) a[t] = 0.0f;
}

// ---------------- fused CGConv edge kernel ----------------
// block = 256 thr (8 wave32), 64 edges. LDS holds Z = [h[dst]|h[src]|ea] as bf16.
// wave nt owns N columns [16*nt, 16*nt+16), computes F = Z@Wf and S = Z@Ws via
// v_wmma_f32_16x16x32_bf16, then atomically scatters sigmoid(F)*softplus(S).
__global__ __launch_bounds__(256) void edge_kernel(
    const float* __restrict__ h, float* __restrict__ out,
    const int* __restrict__ srcI, const int* __restrict__ dstI,
    const float* __restrict__ ea,
    const unsigned short* __restrict__ wF, const unsigned short* __restrict__ wS,
    const float* __restrict__ bF, const float* __restrict__ bS) {

    __shared__ unsigned short zlds[E_TILE * LDS_STRIDE];
    __shared__ int sdst[E_TILE];

    const int tid = threadIdx.x;
    const int ebase = blockIdx.x * E_TILE;
    if (tid < E_TILE) sdst[tid] = dstI[ebase + tid];

    // gather: 72 float4 chunks per edge (32 dst-h, 32 src-h, 8 ea)
    for (int i = tid; i < E_TILE * 72; i += 256) {
        int e = i / 72, q = i - e * 72;
        const float4* p; int col;
        if (q < 32)      { p = (const float4*)&h[(size_t)dstI[ebase + e] * HID] + q;        col = q * 4; }
        else if (q < 64) { p = (const float4*)&h[(size_t)srcI[ebase + e] * HID] + (q - 32); col = HID + (q - 32) * 4; }
        else             { p = (const float4*)&ea[(size_t)(ebase + e) * RBF] + (q - 64);    col = 2 * HID + (q - 64) * 4; }
        float4 v = *p;
        uint2 pk;
        pk.x = (unsigned)f2bf(v.x) | ((unsigned)f2bf(v.y) << 16);
        pk.y = (unsigned)f2bf(v.z) | ((unsigned)f2bf(v.w) << 16);
        *(uint2*)&zlds[e * LDS_STRIDE + col] = pk;
    }
    __syncthreads();

    const int lane = tid & 31;
    const int nt   = tid >> 5;                 // n-tile (wave id), 0..7
    const int half = (lane >> 4) & 1;
    const int nloc = lane & 15;

    v8f accF[4] = {};
    v8f accS[4] = {};

    for (int ks = 0; ks < KSTEPS; ++ks) {
        const int fbase = ((nt * KSTEPS + ks) * 32 + lane) << 4;
        v16bf bFf = *(const v16bf*)&wF[fbase];
        v16bf bSf = *(const v16bf*)&wS[fbase];
        const int kc = ks * 32 + half * 8;
#pragma unroll
        for (int mt = 0; mt < 4; ++mt) {
            const unsigned short* rp = &zlds[(mt * 16 + nloc) * LDS_STRIDE + kc];
            v8bf lo = *(const v8bf*)rp;          // K = kc .. kc+7
            v8bf hi = *(const v8bf*)(rp + 16);   // K = kc+16 .. kc+23
            v16bf a = __builtin_shufflevector(lo, hi,
                0, 1, 2, 3, 4, 5, 6, 7, 8, 9, 10, 11, 12, 13, 14, 15);
            accF[mt] = __builtin_amdgcn_wmma_f32_16x16x32_bf16(
                false, a, false, bFf, (short)0, accF[mt], false, false);
            accS[mt] = __builtin_amdgcn_wmma_f32_16x16x32_bf16(
                false, a, false, bSf, (short)0, accS[mt], false, false);
        }
    }

    const int ncol = nt * 16 + nloc;
    const float biasF = bF[ncol];
    const float biasS = bS[ncol];
#pragma unroll
    for (int mt = 0; mt < 4; ++mt) {
#pragma unroll
        for (int j = 0; j < 8; ++j) {
            int row = mt * 16 + half * 8 + j;        // C layout: M = j + 8*half
            float f = accF[mt][j] + biasF;
            float s = accS[mt][j] + biasS;
            float sig = 1.0f / (1.0f + __expf(-f));
            float sp  = fmaxf(s, 0.0f) + __logf(1.0f + __expf(-fabsf(s)));
            atomicAdd(&out[(size_t)sdst[row] * HID + ncol], sig * sp);
        }
    }
}

// ---------------- pooling + final linear ----------------
__global__ __launch_bounds__(256) void pool_kernel(const float* __restrict__ h,
                                                   const int* __restrict__ batch,
                                                   float* __restrict__ sums,
                                                   float* __restrict__ cnt) {
    int t = blockIdx.x * 256 + threadIdx.x;          // < N_NODES*HID
    int n = t >> 7, c = t & 127;
    int b = batch[n];
    atomicAdd(&sums[b * HID + c], h[t]);
    if (c == 0) atomicAdd(&cnt[b], 1.0f);
}

__global__ __launch_bounds__(128) void final_kernel(const float* __restrict__ sums,
                                                    const float* __restrict__ cnt,
                                                    const float* __restrict__ Wlin,
                                                    const float* __restrict__ blin,
                                                    float* __restrict__ out) {
    __shared__ float prow[HID];
    int g = blockIdx.x, c = threadIdx.x;
    prow[c] = sums[g * HID + c] / fmaxf(cnt[g], 1.0f);
    __syncthreads();
    float acc = blin[c];
#pragma unroll 8
    for (int k = 0; k < HID; ++k) acc = fmaf(prow[k], Wlin[k * HID + c], acc);
    out[g * HID + c] = acc;
}

// ---------------- launcher ----------------
extern "C" void kernel_launch(void* const* d_in, const int* in_sizes, int n_in,
                              void* d_out, int out_size, void* d_ws, size_t ws_size,
                              hipStream_t stream) {
    const int*   x     = (const int*)d_in[0];
    const int*   eidx  = (const int*)d_in[1];
    const int*   srcI  = eidx;
    const int*   dstI  = eidx + N_EDGES;
    const float* ea    = (const float*)d_in[2];
    const int*   batch = (const int*)d_in[3];
    const float* emb   = (const float*)d_in[4];
    const float* Wf[3] = {(const float*)d_in[5],  (const float*)d_in[9],  (const float*)d_in[13]};
    const float* bf[3] = {(const float*)d_in[6],  (const float*)d_in[10], (const float*)d_in[14]};
    const float* Ws[3] = {(const float*)d_in[7],  (const float*)d_in[11], (const float*)d_in[15]};
    const float* bs[3] = {(const float*)d_in[8],  (const float*)d_in[12], (const float*)d_in[16]};
    const float* Wlin  = (const float*)d_in[17];
    const float* blin  = (const float*)d_in[18];
    float* out = (float*)d_out;

    // workspace layout
    char* ws = (char*)d_ws;
    float* hA = (float*)ws;                                   // N_NODES*HID f32
    float* hB = hA + (size_t)N_NODES * HID;                   // N_NODES*HID f32
    unsigned short* wfrag = (unsigned short*)(hB + (size_t)N_NODES * HID);  // 6*FRAG_ELEMS bf16
    float* sums = (float*)(wfrag + (size_t)6 * FRAG_ELEMS);   // N_GRAPHS*HID
    float* cnt  = sums + (size_t)N_GRAPHS * HID;              // N_GRAPHS

    const int nh_blocks = (N_NODES * HID) / 256;              // 16000
    const int fr_blocks = (FRAG_ELEMS + 255) / 256;           // 144
    const int e_blocks  = N_EDGES / E_TILE;                   // 5000

    // weights -> bf16 fragment layout (layer l: F at 2l, S at 2l+1)
    for (int l = 0; l < 3; ++l) {
        wfrag_kernel<<<fr_blocks, 256, 0, stream>>>(Wf[l], wfrag + (size_t)(2 * l) * FRAG_ELEMS);
        wfrag_kernel<<<fr_blocks, 256, 0, stream>>>(Ws[l], wfrag + (size_t)(2 * l + 1) * FRAG_ELEMS);
    }

    // h0 = emb[x]
    embed_kernel<<<nh_blocks, 256, 0, stream>>>(x, emb, hA);

    float* hin = hA;
    float* hout = hB;
    for (int l = 0; l < 3; ++l) {
        copy_kernel<<<nh_blocks, 256, 0, stream>>>(hin, hout);      // residual seed
        edge_kernel<<<e_blocks, 256, 0, stream>>>(
            hin, hout, srcI, dstI, ea,
            wfrag + (size_t)(2 * l) * FRAG_ELEMS,
            wfrag + (size_t)(2 * l + 1) * FRAG_ELEMS,
            bf[l], bs[l]);
        relu_kernel<<<nh_blocks, 256, 0, stream>>>(hout);
        float* t = hin; hin = hout; hout = t;                       // hin now holds layer output
    }

    // global mean pool
    zero_kernel<<<(N_GRAPHS * HID + 255) / 256, 256, 0, stream>>>(sums, N_GRAPHS * HID);
    zero_kernel<<<(N_GRAPHS + 255) / 256, 256, 0, stream>>>(cnt, N_GRAPHS);
    pool_kernel<<<nh_blocks, 256, 0, stream>>>(hin, batch, sums, cnt);

    // final linear [G,128] @ [128,128] + b
    final_kernel<<<N_GRAPHS, 128, 0, stream>>>(sums, cnt, Wlin, blin, out);
}
